// SimTSC_2173253452192
// MI455X (gfx1250) — matrix operand order repaired and checked by reading, not compiled, not run on hardware
//
#include <hip/hip_runtime.h>
#include <math.h>

#define N_NODES 8192
#define FEATD   1024
#define NCLS    64
#define TOPK    32

typedef __attribute__((ext_vector_type(4)))  __bf16 v4bf;
typedef __attribute__((ext_vector_type(8)))  __bf16 v8bf;
typedef __attribute__((ext_vector_type(16))) __bf16 v16bf;
typedef __attribute__((ext_vector_type(8)))  float  v8f;

#define CAT16(lo8, hi8) __builtin_shufflevector(lo8, hi8, 0,1,2,3,4,5,6,7,8,9,10,11,12,13,14,15)
#define WMMA_BF16(A, B, C) \
    __builtin_amdgcn_wmma_f32_16x16x32_bf16(false, (A), false, (B), (short)0, (C), false, false)

// CDNA5 async global->LDS copy (ASYNCcnt-tracked), 16 bytes per lane.
__device__ __forceinline__ void async_copy16(unsigned lds_off, const void* gsrc) {
    asm volatile("global_load_async_to_lds_b128 %0, %1, off"
                 :: "v"(lds_off), "v"(gsrc) : "memory");
}
__device__ __forceinline__ unsigned lds_off(const void* p) {
    return (unsigned)(unsigned long long)p;   // low 32 bits of flat addr = LDS offset
}

// ---------------------------------------------------------------------------
// Kernel 0a: elementwise f32 -> (bf16 hi, bf16 lo), row-major.
// ---------------------------------------------------------------------------
__global__ __launch_bounds__(256) void cvt_hilo_kernel(
    const float* __restrict__ in, __bf16* __restrict__ hi, __bf16* __restrict__ lo, int n4)
{
    const int i = blockIdx.x * 256 + threadIdx.x;
    if (i >= n4) return;
    const float4 v = ((const float4*)in)[i];
    v4bf h, l;
    h[0] = (__bf16)v.x; l[0] = (__bf16)(v.x - (float)h[0]);
    h[1] = (__bf16)v.y; l[1] = (__bf16)(v.y - (float)h[1]);
    h[2] = (__bf16)v.z; l[2] = (__bf16)(v.z - (float)h[2]);
    h[3] = (__bf16)v.w; l[3] = (__bf16)(v.w - (float)h[3]);
    ((v4bf*)hi)[i] = h;
    ((v4bf*)lo)[i] = l;
}

// ---------------------------------------------------------------------------
// Kernel 0b: W[R][C] f32 -> transposed (bf16 hi, lo) [C][R].
// ---------------------------------------------------------------------------
__global__ __launch_bounds__(256) void cvt_transpose_kernel(
    const float* __restrict__ in, __bf16* __restrict__ th, __bf16* __restrict__ tl,
    int R, int C)
{
    const int i = blockIdx.x * 256 + threadIdx.x;
    if (i >= R * C) return;
    const int r = i / C, c = i % C;           // coalesced read along c
    const float v = in[i];
    const __bf16 h = (__bf16)v;
    const __bf16 l = (__bf16)(v - (float)h);
    th[(size_t)c * R + r] = h;
    tl[(size_t)c * R + r] = l;
}

// ---------------------------------------------------------------------------
// Kernel 1: per-row top-32 smallest of adj + normalized exp(-alpha*d) weights.
// One wave32 per row, 8 waves per 256-thread block.
// ---------------------------------------------------------------------------
__global__ __launch_bounds__(256) void topk_kernel(
    const float* __restrict__ adj, const float* __restrict__ alpha,
    int* __restrict__ idx_out, float* __restrict__ w_out)
{
    __shared__ float          cv[8 * 1056];
    __shared__ unsigned short ci[8 * 1056];
    __shared__ float selv[8][TOPK];
    __shared__ int   seli[8][TOPK];

    const int wave = threadIdx.x >> 5;
    const int lane = threadIdx.x & 31;
    const int row  = blockIdx.x * 8 + wave;
    const float* __restrict__ arow = adj + (size_t)row * N_NODES;

    float*          wcv = cv + wave * 1056;
    unsigned short* wci = ci + wave * 1056;
    const int base = lane * 33;
    const float FINF = __builtin_inff();

    float curmax = -FINF; int maxslot = 0;
#pragma unroll
    for (int t = 0; t < 32; ++t) {
        const int j = lane + (t << 5);
        const float v = arow[j];
        wcv[base + t] = v;
        wci[base + t] = (unsigned short)j;
        if (v > curmax) { curmax = v; maxslot = t; }
    }
    for (int t = 32; t < 256; ++t) {
        const int j = lane + (t << 5);
        const float v = arow[j];
        if (v < curmax) {
            wcv[base + maxslot] = v;
            wci[base + maxslot] = (unsigned short)j;
            float m = -FINF; int ms = 0;
#pragma unroll
            for (int s = 0; s < 32; ++s) {
                const float u = wcv[base + s];
                if (u > m) { m = u; ms = s; }
            }
            curmax = m; maxslot = ms;
        }
    }

    for (int sel = 0; sel < TOPK; ++sel) {
        float bestv = FINF; int besta = base;
#pragma unroll
        for (int m = 0; m < 32; ++m) {
            const int a = m * 33 + lane;
            const float u = wcv[a];
            if (u < bestv) { bestv = u; besta = a; }
        }
#pragma unroll
        for (int off = 16; off > 0; off >>= 1) {
            const float ov = __shfl_xor(bestv, off, 32);
            const int   oa = __shfl_xor(besta, off, 32);
            if (ov < bestv) { bestv = ov; besta = oa; }
        }
        if (lane == 0) {
            selv[wave][sel] = bestv;
            seli[wave][sel] = (int)wci[besta];
            wcv[besta] = FINF;
        }
    }

    const float a0 = alpha[0];
    const float e  = expf(-a0 * selv[wave][lane]);
    float s = e;
#pragma unroll
    for (int off = 16; off > 0; off >>= 1) s += __shfl_xor(s, off, 32);
    w_out[(size_t)row * TOPK + lane]   = e / s;
    idx_out[(size_t)row * TOPK + lane] = seli[wave][lane];
}

// ---------------------------------------------------------------------------
// Kernel 2: C[M,N](f32) = A @ B via bf16 hi/lo split (3x v_wmma_f32_16x16x32_bf16).
// A row-major bf16 pairs (Ah, Al) [M][K]; B pre-transposed (Bth, Btl) [N][K].
// Block = 256 thr (8 waves) -> 128x64 tile; wave -> 16x64 strip.
// Double-buffered ASYNCcnt pipeline: stage(next); s_wait_asynccnt<=6; compute(cur).
// ---------------------------------------------------------------------------
#define KC   32   // K elements staged per chunk
#define ASTR 40   // LDS row pitch in bf16 (80 B): conflict-free b128 reads, 16B aligned

__device__ __forceinline__ void stage_tiles(
    __bf16* __restrict__ dAh, __bf16* __restrict__ dAl,
    __bf16* __restrict__ dBh, __bf16* __restrict__ dBl,
    const __bf16* __restrict__ Ah, const __bf16* __restrict__ Al,
    const __bf16* __restrict__ Bth, const __bf16* __restrict__ Btl,
    int tid, int r0, int n0, int K, int kc)
{
    // A hi/lo: 128 rows x 32 k -> 4 x 16B chunks per row  (2 iters x 256 thr)
#pragma unroll
    for (int i = 0; i < 2; ++i) {
        const int t8  = tid + 256 * i;            // 0..511
        const int row = t8 >> 2, ch = t8 & 3;
        const size_t goff = (size_t)(r0 + row) * K + kc + ch * 8;
        async_copy16(lds_off(&dAh[row * ASTR + ch * 8]), Ah + goff);
        async_copy16(lds_off(&dAl[row * ASTR + ch * 8]), Al + goff);
    }
    // B hi/lo: 64 n-rows x 32 k -> 4 chunks per row  (1 iter x 256 thr)
    {
        const int row = tid >> 2, ch = tid & 3;
        const size_t goff = (size_t)(n0 + row) * K + kc + ch * 8;
        async_copy16(lds_off(&dBh[row * ASTR + ch * 8]), Bth + goff);
        async_copy16(lds_off(&dBl[row * ASTR + ch * 8]), Btl + goff);
    }
    // 6 async copies issued per thread per chunk
}

__global__ __launch_bounds__(256) void gemm_bf16x3(
    const __bf16* __restrict__ Ah, const __bf16* __restrict__ Al,
    const __bf16* __restrict__ Bth, const __bf16* __restrict__ Btl,
    float* __restrict__ C, int M, int N, int K)
{
    __shared__ __bf16 sAh[2][128 * ASTR];
    __shared__ __bf16 sAl[2][128 * ASTR];
    __shared__ __bf16 sBh[2][64 * ASTR];
    __shared__ __bf16 sBl[2][64 * ASTR];

    const int tid  = threadIdx.x;
    const int wave = tid >> 5;
    const int lane = tid & 31;
    const int r0   = blockIdx.x * 128;
    const int n0   = blockIdx.y * 64;
    const int lrow = lane & 15;
    const int half = lane >> 4;

    v8f acc[4];
#pragma unroll
    for (int t = 0; t < 4; ++t) acc[t] = (v8f){0.f,0.f,0.f,0.f,0.f,0.f,0.f,0.f};

    // prologue: stage chunk 0 into buffer 0
    stage_tiles(sAh[0], sAl[0], sBh[0], sBl[0], Ah, Al, Bth, Btl, tid, r0, n0, K, 0);

    int p = 0;
    for (int kc = 0; kc < K; kc += KC, p ^= 1) {
        const bool more = (kc + KC) < K;
        if (more) {
            // stage next chunk into the other buffer (freed by trailing barrier
            // of the previous iteration), then wait until only those 6 copies
            // remain outstanding -> current chunk's copies have landed.
            stage_tiles(sAh[p ^ 1], sAl[p ^ 1], sBh[p ^ 1], sBl[p ^ 1],
                        Ah, Al, Bth, Btl, tid, r0, n0, K, kc + KC);
            asm volatile("s_wait_asynccnt 0x6" ::: "memory");
        } else {
            asm volatile("s_wait_asynccnt 0x0" ::: "memory");
        }
        __syncthreads();   // all waves' copies for chunk p landed

        const __bf16* __restrict__ paH = &sAh[p][(wave * 16 + lrow) * ASTR];
        const __bf16* __restrict__ paL = &sAl[p][(wave * 16 + lrow) * ASTR];
        // A frag: lane holds K = {aoff..aoff+15}
        const int aoff = half * 16;
        const v16bf ah = CAT16(*(const v8bf*)(paH + aoff), *(const v8bf*)(paH + aoff + 8));
        const v16bf al = CAT16(*(const v8bf*)(paL + aoff), *(const v8bf*)(paL + aoff + 8));
        const int boff = half * 16;               // B frag: 16 consecutive K at fixed N
#pragma unroll
        for (int t = 0; t < 4; ++t) {
            const __bf16* pbH = &sBh[p][(t * 16 + lrow) * ASTR + boff];
            const __bf16* pbL = &sBl[p][(t * 16 + lrow) * ASTR + boff];
            const v16bf bh = CAT16(*(const v8bf*)pbH, *(const v8bf*)(pbH + 8));
            const v16bf bl = CAT16(*(const v8bf*)pbL, *(const v8bf*)(pbL + 8));
            acc[t] = WMMA_BF16(ah, bh, acc[t]);
            acc[t] = WMMA_BF16(ah, bl, acc[t]);
            acc[t] = WMMA_BF16(al, bh, acc[t]);
        }
        __syncthreads();   // chunk p fully consumed; buffer may be re-staged
    }

    // D layout: VGPR r -> M = r + 8*half, N = lane&15
    const int rbase = r0 + wave * 16 + (half << 3);
#pragma unroll
    for (int t = 0; t < 4; ++t) {
        const int col = n0 + t * 16 + lrow;
#pragma unroll
        for (int r = 0; r < 8; ++r) {
            C[(size_t)(rbase + r) * N + col] = acc[t][r];
        }
    }
}

// ---------------------------------------------------------------------------
// Kernel 3: h1 = relu( spmm(h0) + b1 ), emitted directly as bf16 hi/lo pair.
// ---------------------------------------------------------------------------
__global__ __launch_bounds__(256) void spmm_relu_kernel(
    const float* __restrict__ h, const int* __restrict__ idx,
    const float* __restrict__ w, const float* __restrict__ bias,
    __bf16* __restrict__ outh, __bf16* __restrict__ outl)
{
    __shared__ float lw[TOPK];
    __shared__ int   li[TOPK];
    const int row = blockIdx.x;
    if (threadIdx.x < TOPK) {
        lw[threadIdx.x] = w[(size_t)row * TOPK + threadIdx.x];
        li[threadIdx.x] = idx[(size_t)row * TOPK + threadIdx.x];
    }
    __syncthreads();

    const int f = threadIdx.x << 2;
    float4 acc = {0.f, 0.f, 0.f, 0.f};
#pragma unroll 8
    for (int k = 0; k < TOPK; ++k) {
        const float wk = lw[k];
        const float4 hv = *(const float4*)(h + (size_t)li[k] * FEATD + f);
        acc.x += wk * hv.x; acc.y += wk * hv.y;
        acc.z += wk * hv.z; acc.w += wk * hv.w;
    }
    const float4 b = *(const float4*)(bias + f);
    acc.x = fmaxf(acc.x + b.x, 0.f);
    acc.y = fmaxf(acc.y + b.y, 0.f);
    acc.z = fmaxf(acc.z + b.z, 0.f);
    acc.w = fmaxf(acc.w + b.w, 0.f);

    v4bf hh, hl;
    hh[0] = (__bf16)acc.x; hl[0] = (__bf16)(acc.x - (float)hh[0]);
    hh[1] = (__bf16)acc.y; hl[1] = (__bf16)(acc.y - (float)hh[1]);
    hh[2] = (__bf16)acc.z; hl[2] = (__bf16)(acc.z - (float)hh[2]);
    hh[3] = (__bf16)acc.w; hl[3] = (__bf16)(acc.w - (float)hh[3]);
    *(v4bf*)(outh + (size_t)row * FEATD + f) = hh;
    *(v4bf*)(outl + (size_t)row * FEATD + f) = hl;
}

// ---------------------------------------------------------------------------
// Kernel 4: out = log_softmax( spmm(g) + b2 ). One wave per row.
// ---------------------------------------------------------------------------
__global__ __launch_bounds__(256) void spmm_logsoftmax_kernel(
    const float* __restrict__ g, const int* __restrict__ idx,
    const float* __restrict__ w, const float* __restrict__ b2,
    float* __restrict__ out)
{
    const int wave = threadIdx.x >> 5;
    const int lane = threadIdx.x & 31;
    const int row  = blockIdx.x * 8 + wave;

    const float wv = w[(size_t)row * TOPK + lane];
    const int   ri = idx[(size_t)row * TOPK + lane];

    float acc0 = 0.f, acc1 = 0.f;
#pragma unroll
    for (int k = 0; k < TOPK; ++k) {
        const float wk = __shfl(wv, k, 32);
        const int   rk = __shfl(ri, k, 32);
        const float* __restrict__ gr = g + (size_t)rk * NCLS;
        acc0 += wk * gr[lane];
        acc1 += wk * gr[lane + 32];
    }
    acc0 += b2[lane];
    acc1 += b2[lane + 32];

    float m = fmaxf(acc0, acc1);
#pragma unroll
    for (int off = 16; off > 0; off >>= 1) m = fmaxf(m, __shfl_xor(m, off, 32));
    float s = expf(acc0 - m) + expf(acc1 - m);
#pragma unroll
    for (int off = 16; off > 0; off >>= 1) s += __shfl_xor(s, off, 32);
    const float lse = m + logf(s);

    out[(size_t)row * NCLS + lane]      = acc0 - lse;
    out[(size_t)row * NCLS + lane + 32] = acc1 - lse;
}

// ---------------------------------------------------------------------------
extern "C" void kernel_launch(void* const* d_in, const int* in_sizes, int n_in,
                              void* d_out, int out_size, void* d_ws, size_t ws_size,
                              hipStream_t stream)
{
    const float* x     = (const float*)d_in[0];
    const float* adj   = (const float*)d_in[1];
    const float* W1    = (const float*)d_in[2];
    const float* b1    = (const float*)d_in[3];
    const float* W2    = (const float*)d_in[4];
    const float* b2    = (const float*)d_in[5];
    const float* alpha = (const float*)d_in[6];
    float* out = (float*)d_out;

    const size_t MB = 1ull << 20;
    char* ws = (char*)d_ws;
    int*    idx  = (int*)   (ws + 0 * MB);     //  1 MB
    float*  w    = (float*) (ws + 1 * MB);     //  1 MB
    float*  h0   = (float*) (ws + 2 * MB);     // 32 MB
    __bf16* xh   = (__bf16*)(ws + 34 * MB);    // 16 MB (reused as h1h)
    __bf16* xl   = (__bf16*)(ws + 50 * MB);    // 16 MB (reused as h1l)
    __bf16* w1th = (__bf16*)(ws + 66 * MB);    //  2 MB
    __bf16* w1tl = (__bf16*)(ws + 68 * MB);    //  2 MB
    __bf16* w2th = (__bf16*)(ws + 70 * MB);    //  128 KB
    __bf16* w2tl = (__bf16*)(ws + 70 * MB + 256 * 1024);
    float*  g    = (float*) (ws + 71 * MB);    //  2 MB

    // precision-split conversions (once per element)
    cvt_hilo_kernel<<<(N_NODES * FEATD / 4 + 255) / 256, 256, 0, stream>>>(
        x, xh, xl, N_NODES * FEATD / 4);
    cvt_transpose_kernel<<<(FEATD * FEATD + 255) / 256, 256, 0, stream>>>(
        W1, w1th, w1tl, FEATD, FEATD);
    cvt_transpose_kernel<<<(FEATD * NCLS + 255) / 256, 256, 0, stream>>>(
        W2, w2th, w2tl, FEATD, NCLS);

    topk_kernel<<<N_NODES / 8, 256, 0, stream>>>(adj, alpha, idx, w);

    gemm_bf16x3<<<dim3(N_NODES / 128, FEATD / 64), 256, 0, stream>>>(
        xh, xl, w1th, w1tl, h0, N_NODES, FEATD, FEATD);

    // h1 hi/lo overwrite xh/xl (x is dead after GEMM1)
    __bf16* h1h = xh;
    __bf16* h1l = xl;
    spmm_relu_kernel<<<N_NODES, 256, 0, stream>>>(h0, idx, w, b1, h1h, h1l);

    gemm_bf16x3<<<dim3(N_NODES / 128, NCLS / 64), 256, 0, stream>>>(
        h1h, h1l, w2th, w2tl, g, N_NODES, NCLS, FEATD);

    spmm_logsoftmax_kernel<<<N_NODES / 8, 256, 0, stream>>>(g, idx, w, b2, out);
}
